// ResidualVectorQuantizer_49074296324714
// MI455X (gfx1250) — matrix operand order repaired and checked by reading, not compiled, not run on hardware
//
#include <hip/hip_runtime.h>
#include <stdint.h>

// Residual VQ with Sinkhorn assignment + EMA codebook update, CDNA5/gfx1250.
// Heavy op: per-stage [32768x256] @ [1024x256]^T cosine GEMM -> v_wmma_f32_16x16x32_bf16,
// with double-buffered GLOBAL_LOAD_ASYNC_TO_LDS_B128 tile staging (ASYNCcnt-tracked).
// Sinkhorn collapsed to rank-1 scaling updates (argmax-invariant simplification).

#define DEV __device__ __forceinline__

constexpr int N = 32768, D = 256, M = 4, K = 1024;
constexpr float TAU = 0.2f, SK_EPS = 0.005f, DECAY = 0.99f, LEPS = 1e-5f;

typedef __attribute__((ext_vector_type(16))) __bf16 v16bf;
typedef __attribute__((ext_vector_type(8)))  float  v8f;

struct Frag32 { uint4 lo, hi; };   // 32 bytes == one v16bf operand

// ---- gfx1250 async global->LDS (guarded; falls back to load+ds_store) ----
#if defined(__has_builtin)
#if __has_builtin(__builtin_amdgcn_global_load_async_to_lds_b128) && \
    __has_builtin(__builtin_amdgcn_s_wait_asynccnt)
#define RVQ_ASYNC_LDS 1
#endif
#endif
#ifndef RVQ_ASYNC_LDS
#define RVQ_ASYNC_LDS 0
#endif

#if RVQ_ASYNC_LDS
typedef int rvq_v4i __attribute__((__vector_size__(16)));
DEV void async_cp16(const void* g, void* l) {
  __builtin_amdgcn_global_load_async_to_lds_b128(
      (__attribute__((address_space(1))) rvq_v4i*)g,   // global src (per-lane)
      (__attribute__((address_space(3))) rvq_v4i*)l,   // LDS dst (per-lane)
      0, 0);
}
#endif

// ---- helpers -------------------------------------------------------------
DEV unsigned fkey(float f) {               // monotonic float -> uint key
  unsigned b = __float_as_uint(f);
  return (b & 0x80000000u) ? ~b : (b | 0x80000000u);
}
DEV float funkey(unsigned k) {
  return __uint_as_float((k & 0x80000000u) ? (k & 0x7FFFFFFFu) : ~k);
}
DEV unsigned short f2bf(float f) {         // round-to-nearest-even fp32->bf16
  unsigned b = __float_as_uint(f);
  if ((b & 0x7F800000u) == 0x7F800000u) return (unsigned short)(b >> 16);
  return (unsigned short)((b + 0x7FFFu + ((b >> 16) & 1u)) >> 16);
}
DEV float wave_sum(float v) {
  #pragma unroll
  for (int m = 16; m >= 1; m >>= 1) v += __shfl_xor(v, m, 32);
  return v;
}

// ---- init ----------------------------------------------------------------
__global__ void rvq_init0(const float* __restrict__ x, float* __restrict__ resid,
                          float* __restrict__ xq, float* __restrict__ loss) {
  size_t gid = (size_t)blockIdx.x * 256 + threadIdx.x;   // covers N*D
  resid[gid] = x[gid];
  xq[gid] = 0.f;
  if (gid == 0) loss[0] = 0.f;
}

__global__ void rvq_stage_init(float* __restrict__ u, float* __restrict__ cnt,
                               float* __restrict__ esum, unsigned* __restrict__ scal) {
  size_t gid = (size_t)blockIdx.x * 256 + threadIdx.x;   // covers K*D (>= K)
  if (gid < (size_t)K * D) esum[gid] = 0.f;
  if (gid < (size_t)K) { u[gid] = 0.f; cnt[gid] = 0.f; }
  if (gid == 0) { scal[0] = 0u; scal[1] = 0xFFFFFFFFu; } // max-key / min-key init
}

// ---- row L2-normalize -> bf16 -------------------------------------------
__global__ void rvq_rownorm(const float* __restrict__ src,
                            unsigned short* __restrict__ dst, int rows) {
  int wid = threadIdx.x >> 5, lane = threadIdx.x & 31;
  int row = blockIdx.x * 8 + wid;
  if (row >= rows) return;
  const float* p = src + (size_t)row * D;
  float v[8]; float ss = 0.f;
  #pragma unroll
  for (int j = 0; j < 8; j++) { v[j] = p[lane + 32 * j]; ss += v[j] * v[j]; }
  ss = wave_sum(ss);
  float inv = 1.0f / fmaxf(sqrtf(ss), 1e-12f);
  unsigned short* q = dst + (size_t)row * D;
  #pragma unroll
  for (int j = 0; j < 8; j++) q[lane + 32 * j] = f2bf(v[j] * inv);
}

// ---- WMMA distance GEMM: Dm[n,k] = (1 - xn.en)/TAU, plus global min/max --
__global__ void __launch_bounds__(256)
rvq_dist_gemm(const unsigned short* __restrict__ xn,
              const unsigned short* __restrict__ en,
              float* __restrict__ Dm, unsigned* __restrict__ scal) {
  __shared__ alignas(16) unsigned short sA[2][128][40]; // double-buffered tiles
  __shared__ alignas(16) unsigned short sB[2][128][40];
  __shared__ float smax[8], smin[8];

  int tid = threadIdx.x, wid = tid >> 5, lane = tid & 31;
  int wn = wid >> 2, wk = wid & 3;                // 2x4 wave grid
  int n0 = blockIdx.y * 128, k0 = blockIdx.x * 128;

  v8f acc[4][2];
  #pragma unroll
  for (int a = 0; a < 4; a++)
    #pragma unroll
    for (int b = 0; b < 2; b++) acc[a][b] = (v8f){0,0,0,0,0,0,0,0};

  int lrow = tid >> 1, lhalf = tid & 1;          // 2 threads per tile row
  const unsigned short* gArow = xn + (size_t)(n0 + lrow) * D;
  const unsigned short* gBrow = en + (size_t)(k0 + lrow) * D;
  int rlo = lane & 15, hh = lane >> 4;

  // stage one 128x32 A-tile + 128x32 B-tile into LDS buffer `buf`
  auto stage_tile = [&](int buf, int d0) {
    int col = d0 + lhalf * 16;
#if RVQ_ASYNC_LDS
    async_cp16(gArow + col,     &sA[buf][lrow][lhalf * 16]);
    async_cp16(gArow + col + 8, &sA[buf][lrow][lhalf * 16 + 8]);
    async_cp16(gBrow + col,     &sB[buf][lrow][lhalf * 16]);
    async_cp16(gBrow + col + 8, &sB[buf][lrow][lhalf * 16 + 8]);
#else
    const uint4* ga = (const uint4*)(gArow + col);
    const uint4* gb = (const uint4*)(gBrow + col);
    uint4 a0 = ga[0], a1 = ga[1];
    uint4 b0 = gb[0], b1 = gb[1];
    if (d0 + 32 < D) {                           // gfx1250 global_prefetch_b8
      __builtin_prefetch(ga + 4, 0, 3);
      __builtin_prefetch(gb + 4, 0, 3);
    }
    *(uint4*)&sA[buf][lrow][lhalf * 16]     = a0;
    *(uint4*)&sA[buf][lrow][lhalf * 16 + 8] = a1;
    *(uint4*)&sB[buf][lrow][lhalf * 16]     = b0;
    *(uint4*)&sB[buf][lrow][lhalf * 16 + 8] = b1;
#endif
  };

  stage_tile(0, 0);                              // prologue fill
  for (int it = 0; it < D / 32; ++it) {
#if RVQ_ASYNC_LDS
    __builtin_amdgcn_s_wait_asynccnt(0);         // my async copies landed in LDS
#endif
    __syncthreads();                             // everyone's copies landed
    if (it + 1 < D / 32) stage_tile((it + 1) & 1, (it + 1) * 32);
    int buf = it & 1;

    // A frags (16x32): lane<16 -> row=lane, K{0..7,16..23}; lane>=16 -> K{8..15,24..31}
    v16bf af[4]; v16bf bfr[2];
    #pragma unroll
    for (int mt = 0; mt < 4; mt++) {
      int r = wn * 64 + mt * 16 + rlo;
      Frag32 f;
      f.lo = *(const uint4*)&sA[buf][r][hh * 8];
      f.hi = *(const uint4*)&sA[buf][r][16 + hh * 8];
      af[mt] = __builtin_bit_cast(v16bf, f);
    }
    // B frags (32x16): lane = output column; lane<16 -> K{0..7,16..23}; lane>=16 -> K{8..15,24..31}
    #pragma unroll
    for (int kt = 0; kt < 2; kt++) {
      int ccol = wk * 32 + kt * 16 + rlo;
      Frag32 f;
      f.lo = *(const uint4*)&sB[buf][ccol][hh * 8];
      f.hi = *(const uint4*)&sB[buf][ccol][16 + hh * 8];
      bfr[kt] = __builtin_bit_cast(v16bf, f);
    }
    #pragma unroll
    for (int mt = 0; mt < 4; mt++)
      #pragma unroll
      for (int kt = 0; kt < 2; kt++)
        acc[mt][kt] = __builtin_amdgcn_wmma_f32_16x16x32_bf16(
            false, af[mt], false, bfr[kt], (short)0, acc[mt][kt], false, false);
  }

  // epilogue: d = (1-cos)/TAU, nan->0, store, track min/max
  float tmax = -3.4e38f, tmin = 3.4e38f;
  #pragma unroll
  for (int mt = 0; mt < 4; mt++) {
    #pragma unroll
    for (int kt = 0; kt < 2; kt++) {
      int kcol  = k0 + wk * 32 + kt * 16 + rlo;
      int nbase = n0 + wn * 64 + mt * 16 + hh * 8;
      #pragma unroll
      for (int j = 0; j < 8; j++) {
        float cs = acc[mt][kt][j];
        float dd = (1.0f - cs) * (1.0f / TAU);
        if (dd != dd) dd = 0.0f;
        Dm[(size_t)(nbase + j) * K + kcol] = dd;
        tmax = fmaxf(tmax, dd); tmin = fminf(tmin, dd);
      }
    }
  }
  #pragma unroll
  for (int m2 = 16; m2 >= 1; m2 >>= 1) {
    tmax = fmaxf(tmax, __shfl_xor(tmax, m2, 32));
    tmin = fminf(tmin, __shfl_xor(tmin, m2, 32));
  }
  if (lane == 0) { smax[wid] = tmax; smin[wid] = tmin; }
  __syncthreads();
  if (tid == 0) {
    float bm = smax[0], bn = smin[0];
    for (int w = 1; w < 8; w++) { bm = fmaxf(bm, smax[w]); bn = fminf(bn, smin[w]); }
    atomicMax(&scal[0], fkey(bm));
    atomicMin(&scal[1], fkey(bn));
  }
}

// ---- centering scalars ---------------------------------------------------
__global__ void rvq_finalize_minmax(unsigned* __restrict__ scal) {
  float mx = funkey(scal[0]), mn = funkey(scal[1]);
  float mid = (mx + mn) * 0.5f;
  float amp = fmaxf(mx - mid + 1e-5f, 1e-5f);
  ((float*)scal)[2] = mid;
  ((float*)scal)[3] = 1.0f / (amp * SK_EPS);   // E = exp((mid-d)*inv)
}

// ---- Sinkhorn passes -----------------------------------------------------
// u_k += sum_n E[n,k]*c_n over a 64-row slab. First pass fuses E=exp(...) with c==1.
template <bool DO_EXP>
__global__ void rvq_colsum(float* __restrict__ E, const float* __restrict__ c,
                           float* __restrict__ u, const unsigned* __restrict__ scal) {
  __shared__ float sc[64];
  int tidx = threadIdx.x;
  int nbeg = blockIdx.y * 64;
  if (!DO_EXP && tidx < 64) sc[tidx] = c[nbeg + tidx];
  __syncthreads();
  int k = blockIdx.x * 256 + tidx;
  float mid = ((const float*)scal)[2], inv = ((const float*)scal)[3];
  float s = 0.f;
  for (int j = 0; j < 64; j++) {
    size_t o = (size_t)(nbeg + j) * K + k;
    if (DO_EXP) { float dd = E[o]; float e = __expf((mid - dd) * inv); E[o] = e; s += e; }
    else        { s += E[o] * sc[j]; }
  }
  atomicAdd(&u[k], s);
}

__global__ void rvq_rdiv(float* __restrict__ u, float* __restrict__ r) {
  int k = blockIdx.x * 256 + threadIdx.x;
  r[k] = 1.0f / (u[k] * (float)K);
  u[k] = 0.f;                                   // ready for next col-sum
}

// c_n = 1/(N * sum_k E[n,k]*r_k); r cached in LDS once per block (8 rows reuse it)
__global__ void rvq_rowscale(const float* __restrict__ E, const float* __restrict__ r,
                             float* __restrict__ c) {
  __shared__ float sr[K];
  #pragma unroll
  for (int j = 0; j < K / 256; j++) sr[threadIdx.x + 256 * j] = r[threadIdx.x + 256 * j];
  __syncthreads();
  int wid = threadIdx.x >> 5, lane = threadIdx.x & 31;
  int n = blockIdx.x * 8 + wid;
  const float* row = E + (size_t)n * K;
  float s = 0.f;
  #pragma unroll 4
  for (int j = 0; j < K / 32; j++) { int k = lane + 32 * j; s += row[k] * sr[k]; }
  s = wave_sum(s);
  if (lane == 0) c[n] = 1.0f / (s * (float)N);
}

__global__ void rvq_argmax(const float* __restrict__ E, const float* __restrict__ r,
                           int* __restrict__ idx, float* __restrict__ out_idx,
                           float* __restrict__ cnt, int stage) {
  __shared__ float sr[K];
  #pragma unroll
  for (int j = 0; j < K / 256; j++) sr[threadIdx.x + 256 * j] = r[threadIdx.x + 256 * j];
  __syncthreads();
  int wid = threadIdx.x >> 5, lane = threadIdx.x & 31;
  int n = blockIdx.x * 8 + wid;
  const float* row = E + (size_t)n * K;
  float bv = -3.4e38f; int bk = 0x7FFFFFFF;
  for (int j = 0; j < K / 32; j++) {
    int k = lane + 32 * j;
    float v = row[k] * sr[k];
    if (v > bv || (v == bv && k < bk)) { bv = v; bk = k; }
  }
  #pragma unroll
  for (int m = 16; m >= 1; m >>= 1) {
    float ov = __shfl_xor(bv, m, 32); int ok = __shfl_xor(bk, m, 32);
    if (ov > bv || (ov == bv && ok < bk)) { bv = ov; bk = ok; }
  }
  if (lane == 0) {
    idx[n] = bk;
    out_idx[(size_t)n * M + stage] = (float)bk;
    atomicAdd(&cnt[bk], 1.0f);
  }
}

// ---- residual/quantized update + scatter aggregation ---------------------
__global__ void rvq_update(const float* __restrict__ emb, const int* __restrict__ idx,
                           float* __restrict__ resid, float* __restrict__ xq,
                           float* __restrict__ esum, int stage) {
  size_t gid = (size_t)blockIdx.x * 256 + threadIdx.x;   // covers N*D
  int n = (int)(gid >> 8), d = (int)(gid & 255);
  int k = idx[n];
  float q = emb[((size_t)stage * K + k) * D + d];
  float old = resid[gid];                                // latent into this stage
  atomicAdd(&esum[(size_t)k * D + d], old);              // onehot.T @ latent
  resid[gid] = old - q;
  xq[gid] += q;
}

// ---- EMA codebook update -------------------------------------------------
__global__ void rvq_ema_csize(const float* __restrict__ cs_in, const float* __restrict__ cnt,
                              float* __restrict__ cs_out, int stage) {
  int k = blockIdx.x * 256 + threadIdx.x;
  cs_out[stage * K + k] = cs_in[stage * K + k] * DECAY + cnt[k] * (1.0f - DECAY);
}

__global__ void rvq_ntot(const float* __restrict__ cs_out, unsigned* __restrict__ scal,
                         int stage) {
  __shared__ float sb[256];
  float s = 0.f;
  for (int j = 0; j < K / 256; j++) s += cs_out[stage * K + threadIdx.x + 256 * j];
  sb[threadIdx.x] = s; __syncthreads();
  for (int st = 128; st > 0; st >>= 1) {
    if (threadIdx.x < st) sb[threadIdx.x] += sb[threadIdx.x + st];
    __syncthreads();
  }
  if (threadIdx.x == 0) ((float*)scal)[4] = sb[0];
}

__global__ void rvq_ema_emb(const float* __restrict__ avg_in, const float* __restrict__ esum,
                            const float* __restrict__ cs_out, const unsigned* __restrict__ scal,
                            float* __restrict__ avg_out, float* __restrict__ emb_out,
                            int stage) {
  size_t gid = (size_t)blockIdx.x * 256 + threadIdx.x;   // covers K*D
  size_t o = (size_t)stage * K * D + gid;
  int k = (int)(gid >> 8);
  float na = avg_in[o] * DECAY + esum[gid] * (1.0f - DECAY);
  avg_out[o] = na;
  float nt = ((const float*)scal)[4];
  float w = (cs_out[stage * K + k] + LEPS) / (nt + (float)K * LEPS) * nt;
  emb_out[o] = na / w;
}

// ---- driver --------------------------------------------------------------
extern "C" void kernel_launch(void* const* d_in, const int* in_sizes, int n_in,
                              void* d_out, int out_size, void* d_ws, size_t ws_size,
                              hipStream_t stream) {
  const float* x    = (const float*)d_in[0];
  const float* emb  = (const float*)d_in[1];   // [M,K,D]
  const float* eavg = (const float*)d_in[2];   // [M,K,D]
  const float* csz  = (const float*)d_in[3];   // [M,K]

  float* out_xq   = (float*)d_out;                       // N*D
  float* out_loss = out_xq + (size_t)N * D;               // 1
  float* out_idx  = out_loss + 1;                         // N*M
  float* out_emb  = out_idx + (size_t)N * M;              // M*K*D
  float* out_avg  = out_emb + (size_t)M * K * D;          // M*K*D
  float* out_csz  = out_avg + (size_t)M * K * D;          // M*K

  char* p = (char*)d_ws;
  auto carve = [&](size_t bytes) -> char* {
    char* q = p; p += (bytes + 255) & ~(size_t)255; return q;
  };
  float* resid        = (float*)carve((size_t)N * D * 4);
  unsigned short* xn  = (unsigned short*)carve((size_t)N * D * 2);
  unsigned short* en  = (unsigned short*)carve((size_t)K * D * 2);
  float* E            = (float*)carve((size_t)N * K * 4);     // 128 MB: fits 192 MB L2
  float* r            = (float*)carve((size_t)K * 4);
  float* u            = (float*)carve((size_t)K * 4);
  float* c            = (float*)carve((size_t)N * 4);
  float* cnt          = (float*)carve((size_t)K * 4);
  float* esum         = (float*)carve((size_t)K * D * 4);
  int*   idx          = (int*)carve((size_t)N * 4);
  unsigned* scal      = (unsigned*)carve(256);

  rvq_init0<<<(N * D) / 256, 256, 0, stream>>>(x, resid, out_xq, out_loss);

  for (int s = 0; s < M; s++) {
    rvq_stage_init<<<(K * D) / 256, 256, 0, stream>>>(u, cnt, esum, scal);
    rvq_rownorm<<<N / 8, 256, 0, stream>>>(resid, xn, N);
    rvq_rownorm<<<K / 8, 256, 0, stream>>>(emb + (size_t)s * K * D, en, K);

    dim3 gg(K / 128, N / 128);
    rvq_dist_gemm<<<gg, 256, 0, stream>>>(xn, en, E, scal);
    rvq_finalize_minmax<<<1, 1, 0, stream>>>(scal);

    dim3 gc(K / 256, N / 64);
    // iter 1 (fused exp, c==1), iter 2, iter 3 (row-normalize only; rest argmax-invariant)
    rvq_colsum<true ><<<gc, 256, 0, stream>>>(E, c, u, scal);
    rvq_rdiv<<<K / 256, 256, 0, stream>>>(u, r);
    rvq_rowscale<<<N / 8, 256, 0, stream>>>(E, r, c);
    rvq_colsum<false><<<gc, 256, 0, stream>>>(E, c, u, scal);
    rvq_rdiv<<<K / 256, 256, 0, stream>>>(u, r);
    rvq_rowscale<<<N / 8, 256, 0, stream>>>(E, r, c);
    rvq_colsum<false><<<gc, 256, 0, stream>>>(E, c, u, scal);
    rvq_rdiv<<<K / 256, 256, 0, stream>>>(u, r);

    rvq_argmax<<<N / 8, 256, 0, stream>>>(E, r, idx, out_idx, cnt, s);
    rvq_update<<<(N * D) / 256, 256, 0, stream>>>(emb, idx, resid, out_xq, esum, s);

    rvq_ema_csize<<<K / 256, 256, 0, stream>>>(csz, cnt, out_csz, s);
    rvq_ntot<<<1, 256, 0, stream>>>(out_csz, scal, s);
    rvq_ema_emb<<<(K * D) / 256, 256, 0, stream>>>(eavg, esum, out_csz, scal,
                                                   out_avg, out_emb, s);
  }
}
